// EdgeEnhancedAttention_44178033607025
// MI455X (gfx1250) — matrix operand order repaired and checked by reading, not compiled, not run on hardware
//
#include <hip/hip_runtime.h>
#include <hip/hip_bf16.h>
#include <math.h>

typedef __attribute__((ext_vector_type(16))) __bf16 v16bf;
typedef __attribute__((ext_vector_type(8)))  float  v8f;
typedef __attribute__((ext_vector_type(4)))  float  v4f;
typedef __attribute__((ext_vector_type(4)))  unsigned int u32x4;
typedef __attribute__((ext_vector_type(8)))  int  i32x8;
typedef __attribute__((ext_vector_type(4)))  int  i32x4;

#define WAVES 8
#define AWAVES 4
#define SCALE_QK 0.17677669529663687f  // 32^-0.5

union AF { v16bf v; __bf16 e[16]; };
union CF { v8f  v; float  e[8];  };

__device__ __forceinline__ __bf16 bf(float x) { return (__bf16)x; }

#if __has_builtin(__builtin_amdgcn_tensor_load_to_lds)
#define HAVE_TDM 1
#else
#define HAVE_TDM 0
#endif

// ---------------------------------------------------------------------------
// TDM: async-load a 16(row) x 32(col) f32 tile from a row-major tensor with
// row stride 1024 elements into LDS (dense 16x32).  D# per CDNA5 ISA ch.8.
// ---------------------------------------------------------------------------
__device__ __forceinline__ void tdm_issue_tile(const float* gsrc, unsigned lds_off)
{
#if HAVE_TDM
    unsigned long long ga = (unsigned long long)(size_t)gsrc;
    u32x4 g0;
    g0[0] = 1u;                                     // count=1, user mode
    g0[1] = lds_off;                                // lds_addr (bytes)
    g0[2] = (unsigned)ga;                           // global_addr[31:0]
    g0[3] = ((unsigned)(ga >> 32) & 0x01FFFFFFu)    // global_addr[56:32]
            | (2u << 30);                           // type = 2 ("image")
    i32x8 g1;
    g1[0] = (int)(2u << 16);        // workgroup_mask=0 | data_size=2 (4 bytes)
    g1[1] = (int)(1024u << 16);     // barrier_addr=0 | tensor_dim0[15:0]=1024
    g1[2] = (int)(16u << 16);       // tensor_dim0[31:16]=0 | tensor_dim1[15:0]=16
    g1[3] = (int)(32u << 16);       // tensor_dim1[31:16]=0 | tile_dim0=32
    g1[4] = (int)16;                // tile_dim1=16 | tile_dim2=0
    g1[5] = (int)1024;              // tensor_dim0_stride[31:0]=1024
    g1[6] = 0;                      // stride hi | tensor_dim1_stride lo
    g1[7] = 0;
    i32x4 g2 = {0, 0, 0, 0};
    i32x4 g3 = {0, 0, 0, 0};
#if defined(__clang_major__) && (__clang_major__ >= 23)
    i32x8 g4 = {0, 0, 0, 0, 0, 0, 0, 0};
    __builtin_amdgcn_tensor_load_to_lds(g0, g1, g2, g3, g4, 0);
#else
    __builtin_amdgcn_tensor_load_to_lds(g0, g1, g2, g3, 0);
#endif
#else
    (void)gsrc; (void)lds_off;
#endif
}

// Fallback: cooperative vectorized copy global -> LDS (4 x float4 per lane).
__device__ __forceinline__ void copy_tile_fallback(const float* __restrict__ gsrc,
                                                   float* __restrict__ ldst, int lane)
{
#if !HAVE_TDM
#pragma unroll
    for (int j = 0; j < 4; ++j) {
        const int qi  = j * 32 + lane;     // 0..127 quads
        const int row = qi >> 3;
        const int c4  = (qi & 7) * 4;
        const v4f val = __builtin_nontemporal_load((const v4f*)(gsrc + row * 1024 + c4));
        *(v4f*)(ldst + row * 32 + c4) = val;
    }
#else
    (void)gsrc; (void)ldst; (void)lane;
#endif
}

// ---------------------------------------------------------------------------
// Kernel 1: fused QKV projection.  out(4096 x 768) = x(4096 x 256) @ Wcat^T.
// Q,K stored [B,H,N,DK] bf16; V stored transposed [B,H,DK,N] bf16.
// ---------------------------------------------------------------------------
__global__ __launch_bounds__(256) void qkv_kernel(
    const float* __restrict__ x,
    const float* __restrict__ Wq, const float* __restrict__ bq,
    const float* __restrict__ Wk, const float* __restrict__ bk,
    const float* __restrict__ Wv, const float* __restrict__ bv,
    __bf16* __restrict__ Qg, __bf16* __restrict__ Kg, __bf16* __restrict__ Vt)
{
    const int wid  = threadIdx.x >> 5;
    const int lane = threadIdx.x & 31;
    const int task = blockIdx.x * WAVES + wid;     // 256 mtiles * 12 ngroups = 3072
    const int mt   = task / 12;
    const int ng   = task % 12;
    const int ln   = lane & 15;
    const int hi   = lane >> 4;
    const int r0   = hi * 8;

    CF acc[4];
#pragma unroll
    for (int t = 0; t < 4; ++t)
#pragma unroll
        for (int i = 0; i < 8; ++i) acc[t].e[i] = 0.0f;

    for (int kc = 0; kc < 8; ++kc) {
        AF a;
        {
            const float* xr = x + (mt * 16 + ln) * 256 + kc * 32 + hi * 16;
#pragma unroll
            for (int i = 0; i < 8; ++i) a.e[i]     = bf(xr[i]);
#pragma unroll
            for (int i = 0; i < 8; ++i) a.e[8 + i] = bf(xr[8 + i]);
        }
#pragma unroll
        for (int t = 0; t < 4; ++t) {
            const int n  = (ng * 4 + t) * 16 + ln;           // 0..767
            const float* W = (n < 256) ? Wq : (n < 512) ? Wk : Wv;
            const int f  = n & 255;
            const float* wr = W + f * 256 + kc * 32 + hi * 16;
            AF bm;
#pragma unroll
            for (int i = 0; i < 16; ++i) bm.e[i] = bf(wr[i]);
            acc[t].v = __builtin_amdgcn_wmma_f32_16x16x32_bf16(
                false, a.v, false, bm.v, (short)0, acc[t].v, false, false);
        }
    }

#pragma unroll
    for (int t = 0; t < 4; ++t) {
        const int n   = (ng * 4 + t) * 16 + ln;
        const int sel = n >> 8;
        const int f   = n & 255;
        const int h   = f >> 5;
        const int dk  = f & 31;
        const float bias = (sel == 0 ? bq : (sel == 1 ? bk : bv))[f];
#pragma unroll
        for (int r = 0; r < 8; ++r) {
            const int m = mt * 16 + r0 + r;
            const int b = m >> 10, q = m & 1023;
            const float val = acc[t].e[r] + bias;
            if (sel == 0)
                Qg[(((size_t)(b * 8 + h)) * 1024 + q) * 32 + dk] = bf(val);
            else if (sel == 1)
                Kg[(((size_t)(b * 8 + h)) * 1024 + q) * 32 + dk] = bf(val);
            else
                Vt[(((size_t)(b * 8 + h)) * 32 + dk) * 1024 + q] = bf(val);
        }
    }
}

// ---------------------------------------------------------------------------
// Kernel 2: fused bias + mask + flash-softmax attention.
// One wave per (b, h, 16-query tile); 64 key chunks of 32.
// Bias tensors streamed global->LDS by the Tensor Data Mover, double-buffered.
// ---------------------------------------------------------------------------
struct alignas(16) AttnSmem {
    float  bias[AWAVES][2][3][16 * 32];  // TDM destination, 2-deep pipeline
    float  s[AWAVES][16 * 33];           // score transpose tile (pad 33)
    __bf16 p[AWAVES][16 * 32];           // softmax probs (bf16 A-frag source)
    float  a[AWAVES][16];                // per-row alpha / inv-sum
};

__global__ __launch_bounds__(128) void attn_kernel(
    const __bf16* __restrict__ Qg, const __bf16* __restrict__ Kg,
    const __bf16* __restrict__ Vt,
    const float* __restrict__ ebias, const float* __restrict__ sbias,
    const float* __restrict__ rbias,
    const unsigned char* __restrict__ mask,
    __bf16* __restrict__ AO)
{
    __shared__ AttnSmem sm;

    const int wid  = threadIdx.x >> 5;
    const int lane = threadIdx.x & 31;
    const int task = blockIdx.x * AWAVES + wid;   // 4*8*64 = 2048
    const int qt = task & 63;
    const int h  = (task >> 6) & 7;
    const int b  = task >> 9;
    const int q0 = qt * 16;
    const int ln = lane & 15;
    const int hi = lane >> 4;
    const int r0 = hi * 8;

    // wave-uniform (SGPR) copies for descriptor building
    const int uwid  = __builtin_amdgcn_readfirstlane(wid);
    const int utask = __builtin_amdgcn_readfirstlane(task);
    const int ub  = utask >> 9;
    const int uh  = (utask >> 6) & 7;
    const int uq0 = (utask & 63) * 16;
    const int ubiasOff = ((ub * 8 + uh) << 20) + (uq0 << 10);  // (b*8+h)*1M + q0*1024

    const size_t bh = (size_t)(b * 8 + h);
    const __bf16* Qp = Qg + bh * 1024 * 32;
    const __bf16* Kp = Kg + bh * 1024 * 32;
    const __bf16* Vp = Vt + bh * 32 * 1024;
    const float* bsrc0 = ebias + ubiasOff;
    const float* bsrc1 = sbias + ubiasOff;
    const float* bsrc2 = rbias + ubiasOff;

    // Q A-fragment (persistent)
    AF qa;
    {
        const __bf16* src = Qp + (q0 + ln) * 32 + hi * 16;
#pragma unroll
        for (int i = 0; i < 8; ++i) { qa.e[i] = src[i]; qa.e[8 + i] = src[8 + i]; }
    }

    CF acc0, acc1, zf;
#pragma unroll
    for (int i = 0; i < 8; ++i) { acc0.e[i] = 0.f; acc1.e[i] = 0.f; zf.e[i] = 0.f; }
    float mrow = -INFINITY, srow = 0.0f;   // live in lanes 0..15 (row = ln)

    // prologue: stage chunk 0 biases into buffer 0
#if HAVE_TDM
    {
        const unsigned o0 = (unsigned)((char*)&sm.bias[uwid][0][0][0] - (char*)&sm);
        tdm_issue_tile(bsrc0, o0);
        tdm_issue_tile(bsrc1, o0 + 2048);
        tdm_issue_tile(bsrc2, o0 + 4096);
    }
#else
    copy_tile_fallback(bsrc0, &sm.bias[wid][0][0][0], lane);
    copy_tile_fallback(bsrc1, &sm.bias[wid][0][1][0], lane);
    copy_tile_fallback(bsrc2, &sm.bias[wid][0][2][0], lane);
#endif

    int buf = 0;
    for (int kc = 0; kc < 64; ++kc) {
        const int kb = kc * 32;

        // stage chunk kc+1 into the other buffer, then wait for chunk kc's tiles
        if (kc + 1 < 64) {
#if HAVE_TDM
            const unsigned o1 = (unsigned)((char*)&sm.bias[uwid][buf ^ 1][0][0] - (char*)&sm);
            tdm_issue_tile(bsrc0 + kb + 32, o1);
            tdm_issue_tile(bsrc1 + kb + 32, o1 + 2048);
            tdm_issue_tile(bsrc2 + kb + 32, o1 + 4096);
            __builtin_amdgcn_s_wait_tensorcnt(3);
#else
            copy_tile_fallback(bsrc0 + kb + 32, &sm.bias[wid][buf ^ 1][0][0], lane);
            copy_tile_fallback(bsrc1 + kb + 32, &sm.bias[wid][buf ^ 1][1][0], lane);
            copy_tile_fallback(bsrc2 + kb + 32, &sm.bias[wid][buf ^ 1][2][0], lane);
            asm volatile("s_wait_dscnt 0x0" ::: "memory");
#endif
        } else {
#if HAVE_TDM
            __builtin_amdgcn_s_wait_tensorcnt(0);
#else
            asm volatile("s_wait_dscnt 0x0" ::: "memory");
#endif
        }

        const float* bb0 = &sm.bias[wid][buf][0][0];
        const float* bb1 = &sm.bias[wid][buf][1][0];
        const float* bb2 = &sm.bias[wid][buf][2][0];

        // ---- scores: 2 x (16q x 16k) WMMA, DK = 32 = one K step ----
#pragma unroll
        for (int t = 0; t < 2; ++t) {
            const int key = kb + t * 16 + ln;
            AF kf;
            {
                const __bf16* src = Kp + key * 32 + hi * 16;
#pragma unroll
                for (int i = 0; i < 16; ++i) kf.e[i] = src[i];
            }
            CF sc;
            sc.v = __builtin_amdgcn_wmma_f32_16x16x32_bf16(
                false, qa.v, false, kf.v, (short)0, zf.v, false, false);

            const bool km = mask[b * 1024 + key] != 0;
#pragma unroll
            for (int r = 0; r < 8; ++r) {
                const int bi = (r0 + r) * 32 + t * 16 + ln;   // LDS tile index
                const float e1 = bb0[bi];
                const float e2 = bb1[bi];
                const float e3 = bb2[bi];
                float v = sc.e[r] * SCALE_QK + e1 + e2 + e3;
                v = km ? -INFINITY : v;
                sm.s[wid][(r0 + r) * 33 + t * 16 + ln] = v;
            }
        }
        asm volatile("s_wait_dscnt 0x0" ::: "memory");

        // ---- online softmax: lanes 0..15, one row each ----
        if (hi == 0) {
            float vb[32];
            float mloc = -INFINITY;
#pragma unroll
            for (int i = 0; i < 32; ++i) {
                vb[i] = sm.s[wid][ln * 33 + i];
                mloc = fmaxf(mloc, vb[i]);
            }
            const float mnew  = fmaxf(mrow, mloc);
            const float alpha = (mrow == -INFINITY) ? 0.0f : __expf(mrow - mnew);
            float sum = 0.0f;
            if (mnew == -INFINITY) {
#pragma unroll
                for (int i = 0; i < 32; ++i) sm.p[wid][ln * 32 + i] = bf(0.0f);
            } else {
#pragma unroll
                for (int i = 0; i < 32; ++i) {
                    const float p = __expf(vb[i] - mnew);
                    sum += p;
                    sm.p[wid][ln * 32 + i] = bf(p);
                }
            }
            srow = srow * alpha + sum;
            mrow = mnew;
            sm.a[wid][ln] = alpha;
        }
        asm volatile("s_wait_dscnt 0x0" ::: "memory");

        // ---- rescale accumulators (per-row alpha via LDS broadcast) ----
#pragma unroll
        for (int r = 0; r < 8; ++r) {
            const float al = sm.a[wid][r0 + r];
            acc0.e[r] *= al;
            acc1.e[r] *= al;
        }

        // ---- P A-fragment from LDS ----
        AF pf;
        {
            const __bf16* src = &sm.p[wid][ln * 32 + hi * 16];
#pragma unroll
            for (int i = 0; i < 8; ++i) { pf.e[i] = src[i]; pf.e[8 + i] = src[8 + i]; }
        }

        // ---- PV: two dk-halves ----
        {
            AF vf;
            const __bf16* src = Vp + (size_t)(0 + ln) * 1024 + kb + hi * 16;
#pragma unroll
            for (int i = 0; i < 16; ++i) vf.e[i] = src[i];
            acc0.v = __builtin_amdgcn_wmma_f32_16x16x32_bf16(
                false, pf.v, false, vf.v, (short)0, acc0.v, false, false);
        }
        {
            AF vf;
            const __bf16* src = Vp + (size_t)(16 + ln) * 1024 + kb + hi * 16;
#pragma unroll
            for (int i = 0; i < 16; ++i) vf.e[i] = src[i];
            acc1.v = __builtin_amdgcn_wmma_f32_16x16x32_bf16(
                false, pf.v, false, vf.v, (short)0, acc1.v, false, false);
        }

        buf ^= 1;
    }

    // ---- finalize: divide by row sums (0 for fully-masked rows == nan_to_num) ----
    if (hi == 0) sm.a[wid][ln] = (srow > 0.0f) ? (1.0f / srow) : 0.0f;
    asm volatile("s_wait_dscnt 0x0" ::: "memory");
#pragma unroll
    for (int r = 0; r < 8; ++r) {
        const float inv = sm.a[wid][r0 + r];
        const int q = q0 + r0 + r;
        const size_t o = ((size_t)b * 1024 + q) * 256 + h * 32;
        AO[o + ln]      = bf(acc0.e[r] * inv);
        AO[o + 16 + ln] = bf(acc1.e[r] * inv);
    }
}

// ---------------------------------------------------------------------------
// Kernel 3: output projection. out(4096 x 256) = AO(bf16) @ Wo^T + bo (f32).
// ---------------------------------------------------------------------------
__global__ __launch_bounds__(256) void oproj_kernel(
    const __bf16* __restrict__ AO, const float* __restrict__ Wo,
    const float* __restrict__ bo, float* __restrict__ out)
{
    const int wid  = threadIdx.x >> 5;
    const int lane = threadIdx.x & 31;
    const int task = blockIdx.x * WAVES + wid;   // 256 * 16 = 4096
    const int mt = task >> 4;
    const int nt = task & 15;
    const int ln = lane & 15;
    const int hi = lane >> 4;
    const int r0 = hi * 8;

    CF acc;
#pragma unroll
    for (int i = 0; i < 8; ++i) acc.e[i] = 0.0f;

    for (int kc = 0; kc < 8; ++kc) {
        AF a;
        {
            const __bf16* src = AO + (size_t)(mt * 16 + ln) * 256 + kc * 32 + hi * 16;
#pragma unroll
            for (int i = 0; i < 8; ++i) { a.e[i] = src[i]; a.e[8 + i] = src[8 + i]; }
        }
        AF bm;
        {
            const int n = nt * 16 + ln;
            const float* wr = Wo + n * 256 + kc * 32 + hi * 16;
#pragma unroll
            for (int i = 0; i < 16; ++i) bm.e[i] = bf(wr[i]);
        }
        acc.v = __builtin_amdgcn_wmma_f32_16x16x32_bf16(
            false, a.v, false, bm.v, (short)0, acc.v, false, false);
    }

    const int n = nt * 16 + ln;
    const float bias = bo[n];
#pragma unroll
    for (int r = 0; r < 8; ++r)
        out[(size_t)(mt * 16 + r0 + r) * 256 + n] = acc.e[r] + bias;
}

// ---------------------------------------------------------------------------
extern "C" void kernel_launch(void* const* d_in, const int* in_sizes, int n_in,
                              void* d_out, int out_size, void* d_ws, size_t ws_size,
                              hipStream_t stream)
{
    const float* x     = (const float*)d_in[0];
    const float* ebias = (const float*)d_in[1];
    const float* sbias = (const float*)d_in[2];
    const float* rbias = (const float*)d_in[3];
    const unsigned char* mask = (const unsigned char*)d_in[4];
    const float* Wq = (const float*)d_in[5];
    const float* bq = (const float*)d_in[6];
    const float* Wk = (const float*)d_in[7];
    const float* bk = (const float*)d_in[8];
    const float* Wv = (const float*)d_in[9];
    const float* bv = (const float*)d_in[10];
    const float* Wo = (const float*)d_in[11];
    const float* bo = (const float*)d_in[12];

    const size_t QKV_ELEMS = (size_t)4 * 8 * 1024 * 32;  // 1M bf16 each
    __bf16* Qg = (__bf16*)d_ws;
    __bf16* Kg = Qg + QKV_ELEMS;
    __bf16* Vt = Kg + QKV_ELEMS;
    __bf16* AO = Vt + QKV_ELEMS;

    qkv_kernel<<<384, 256, 0, stream>>>(x, Wq, bq, Wk, bk, Wv, bv, Qg, Kg, Vt);
    attn_kernel<<<512, 128, 0, stream>>>(Qg, Kg, Vt, ebias, sbias, rbias, mask, AO);
    oproj_kernel<<<512, 256, 0, stream>>>(AO, Wo, bo, (float*)d_out);
}
